// GraphAttentionLayer_12051678232592
// MI455X (gfx1250) — matrix operand order repaired and checked by reading, not compiled, not run on hardware
//
#include <hip/hip_runtime.h>
#include <hip/hip_fp16.h>

typedef __attribute__((ext_vector_type(16))) _Float16 v16h;
typedef __attribute__((ext_vector_type(8)))  float    v8f;

#define GAT_B    4
#define GAT_N    2048
#define GAT_FIN  128
#define GAT_FOUT 64
#define GAT_H    4
#define GAT_ALPHA 0.2f

// ---------------------------------------------------------------------------
// Stage 1: ht = einsum('hio,bni->bnho'), e_src/e_dst, and f16 B-fragment
// swizzle of ht for the stage-2 WMMA (B matrix 32(K=j) x 16(N=o) tiles:
// lane = (o&15) + ((j&31)>=16)*16 ; per-lane 16 consecutive K halves).
// ---------------------------------------------------------------------------
__global__ __launch_bounds__(256) void gat_stage1(
    const float* __restrict__ hin, const float* __restrict__ W,
    const float* __restrict__ a, _Float16* __restrict__ ht16,
    float* __restrict__ es, float* __restrict__ ed)
{
    __shared__ float s_h[GAT_FIN];
    __shared__ float s_es[GAT_H];
    __shared__ float s_ed[GAT_H];

    const int n = blockIdx.x;
    const int b = blockIdx.y;
    const int t = threadIdx.x;
    const int hh = t >> 6;     // head 0..3
    const int o  = t & 63;     // out feature 0..63

    if (t < GAT_FIN) s_h[t] = hin[((size_t)b * GAT_N + n) * GAT_FIN + t];
    if (t < GAT_H) { s_es[t] = 0.f; s_ed[t] = 0.f; }
    __syncthreads();

    const float* wp = W + (size_t)hh * GAT_FIN * GAT_FOUT + o;
    float acc = 0.f;
#pragma unroll 8
    for (int k = 0; k < GAT_FIN; ++k)
        acc = fmaf(s_h[k], wp[(size_t)k * GAT_FOUT], acc);

    // attention logit partial sums (a: (H, 2*FOUT, 1))
    const float asrc = a[hh * 2 * GAT_FOUT + o];
    const float adst = a[hh * 2 * GAT_FOUT + GAT_FOUT + o];
    atomicAdd(&s_es[hh], acc * asrc);
    atomicAdd(&s_ed[hh], acc * adst);

    // swizzled f16 store: fragment = 32x16 (j x o) B tile for WMMA
    const int kr   = n & 31;
    const int lane = (o & 15) + ((kr >> 4) << 4);
    const int idx  = kr & 15;
    const int frag = ((b * GAT_H + hh) * (GAT_N / 32) + (n >> 5)) * 4 + (o >> 4);
    ht16[(size_t)frag * 512 + lane * 16 + idx] = (_Float16)acc;

    __syncthreads();
    if (t < GAT_H) {
        es[((size_t)b * GAT_H + t) * GAT_N + n] = s_es[t];
        ed[((size_t)b * GAT_H + t) * GAT_N + n] = s_ed[t];
    }
}

// ---------------------------------------------------------------------------
// Stage 2: fused masked softmax + att @ ht via v_wmma_f32_16x16x32_f16.
// Block = 4 waves; block owns (b, h, 16-row tile); wave w owns j-slice of 512.
// ---------------------------------------------------------------------------
__device__ __forceinline__ float gat_logit(float es_i, float edv, float adjv)
{
    float e = es_i + edv;
    e = (e >= 0.f) ? e : GAT_ALPHA * e;
    return (adjv != 0.f) ? e : (e - 1e20f);   // mask: -1e20*(1-adj)
}

__global__ __launch_bounds__(128) void gat_stage2(
    const float* __restrict__ adj, const _Float16* __restrict__ ht16,
    const float* __restrict__ es, const float* __restrict__ ed,
    float* __restrict__ out)
{
    __shared__ float s_acc[4][16][GAT_FOUT];   // 16 KB
    __shared__ float s_m[4][16];
    __shared__ float s_l[4][16];

    const int itile = blockIdx.x;          // 0..127 (row tile of 16)
    const int hh    = blockIdx.y;          // head
    const int b     = blockIdx.z;          // batch
    const int t     = threadIdx.x;
    const int wave  = t >> 5;
    const int lane  = t & 31;
    const int row   = lane & 15;
    const int hilane = lane >> 4;          // 0: K low half, 1: K high half

    const int i0 = itile * 16;
    const int i  = i0 + row;
    const size_t ebase = ((size_t)b * GAT_H + hh) * GAT_N;

    const float  es_i    = es[ebase + i];
    const float* adj_row = adj + (size_t)i * GAT_N;
    const float* ed_p    = ed + ebase;
    const int    j0w     = wave * 512;

    // ---------------- pass 1: per-row max over this wave's j-slice ---------
    float mi = -3.0e38f;
    {
        const int js = j0w + (hilane << 8);   // each half-wave scans 256 j's
        for (int c = 0; c < 64; ++c) {
            const int j = js + c * 4;
            const float4 av = *(const float4*)(adj_row + j);
            const float4 ev = *(const float4*)(ed_p + j);
            mi = fmaxf(mi, gat_logit(es_i, ev.x, av.x));
            mi = fmaxf(mi, gat_logit(es_i, ev.y, av.y));
            mi = fmaxf(mi, gat_logit(es_i, ev.z, av.z));
            mi = fmaxf(mi, gat_logit(es_i, ev.w, av.w));
        }
        mi = fmaxf(mi, __shfl_xor(mi, 16, 32));  // lanes L and L+16 share a row
    }

    // ---------------- pass 2: p = exp(e - m), l sum, WMMA accumulation -----
    v8f acc0 = {}, acc1 = {}, acc2 = {}, acc3 = {};
    float lsum = 0.f;

    const _Float16* fragbase =
        ht16 + ((size_t)(b * GAT_H + hh) * (GAT_N / 32)) * 2048; // 4 frags * 512 halves per jb

    for (int jb = j0w; jb < j0w + 512; jb += 32) {
        const int jA = jb + (hilane << 4);   // A-layout spans for this lane
        const float4 a0 = *(const float4*)(adj_row + jA);
        const float4 a1 = *(const float4*)(adj_row + jA + 4);
        const float4 a2 = *(const float4*)(adj_row + jA + 8);
        const float4 a3 = *(const float4*)(adj_row + jA + 12);
        const float4 e0 = *(const float4*)(ed_p + jA);
        const float4 e1 = *(const float4*)(ed_p + jA + 4);
        const float4 e2 = *(const float4*)(ed_p + jA + 8);
        const float4 e3 = *(const float4*)(ed_p + jA + 12);

        union { v16h v; _Float16 x[16]; } A;
#define GAT_P(slot, av, ev)                                            \
        { float p = __expf(gat_logit(es_i, ev, av) - mi);              \
          lsum += p; A.x[slot] = (_Float16)p; }
        GAT_P(0,  a0.x, e0.x) GAT_P(1,  a0.y, e0.y)
        GAT_P(2,  a0.z, e0.z) GAT_P(3,  a0.w, e0.w)
        GAT_P(4,  a1.x, e1.x) GAT_P(5,  a1.y, e1.y)
        GAT_P(6,  a1.z, e1.z) GAT_P(7,  a1.w, e1.w)
        GAT_P(8,  a2.x, e2.x) GAT_P(9,  a2.y, e2.y)
        GAT_P(10, a2.z, e2.z) GAT_P(11, a2.w, e2.w)
        GAT_P(12, a3.x, e3.x) GAT_P(13, a3.y, e3.y)
        GAT_P(14, a3.z, e3.z) GAT_P(15, a3.w, e3.w)
#undef GAT_P

        // B fragments: per-lane 16 contiguous halves (32B, 32B-aligned)
        const _Float16* fb = fragbase + (size_t)(jb >> 5) * 2048 + lane * 16;
        const v16h b0 = *(const v16h*)(fb);
        const v16h b1 = *(const v16h*)(fb + 512);
        const v16h b2 = *(const v16h*)(fb + 1024);
        const v16h b3 = *(const v16h*)(fb + 1536);

        acc0 = __builtin_amdgcn_wmma_f32_16x16x32_f16(false, A.v, false, b0,
                                                      (short)0, acc0, false, false);
        acc1 = __builtin_amdgcn_wmma_f32_16x16x32_f16(false, A.v, false, b1,
                                                      (short)0, acc1, false, false);
        acc2 = __builtin_amdgcn_wmma_f32_16x16x32_f16(false, A.v, false, b2,
                                                      (short)0, acc2, false, false);
        acc3 = __builtin_amdgcn_wmma_f32_16x16x32_f16(false, A.v, false, b3,
                                                      (short)0, acc3, false, false);
    }
    lsum += __shfl_xor(lsum, 16, 32);   // combine the two half-lanes of a row

    // ---------------- spill per-wave (m, l, acc) to LDS --------------------
#pragma unroll
    for (int r = 0; r < 8; ++r) {
        const int ri = r + hilane * 8;      // C layout: row = r (+8 for hi lanes)
        s_acc[wave][ri][ 0 + row] = acc0[r];
        s_acc[wave][ri][16 + row] = acc1[r];
        s_acc[wave][ri][32 + row] = acc2[r];
        s_acc[wave][ri][48 + row] = acc3[r];
    }
    if (!hilane) { s_m[wave][row] = mi; s_l[wave][row] = lsum; }
    __syncthreads();

    // ---------------- cross-wave flash merge + normalize + store -----------
    const int i2 = t >> 3;            // row 0..15
    const int o0 = (t & 7) * 8;       // 8 contiguous output features
    const float m0 = s_m[0][i2], m1 = s_m[1][i2], m2 = s_m[2][i2], m3 = s_m[3][i2];
    const float ms = fmaxf(fmaxf(m0, m1), fmaxf(m2, m3));
    const float f0 = __expf(m0 - ms), f1 = __expf(m1 - ms);
    const float f2 = __expf(m2 - ms), f3 = __expf(m3 - ms);
    const float L = f0 * s_l[0][i2] + f1 * s_l[1][i2] +
                    f2 * s_l[2][i2] + f3 * s_l[3][i2];
    const float invL = 1.0f / L;

    float* op = out + ((size_t)(b * GAT_N + i0 + i2)) * (GAT_H * GAT_FOUT)
                    + hh * GAT_FOUT + o0;
#pragma unroll
    for (int oo = 0; oo < 8; ++oo) {
        const float v = f0 * s_acc[0][i2][o0 + oo] + f1 * s_acc[1][i2][o0 + oo]
                      + f2 * s_acc[2][i2][o0 + oo] + f3 * s_acc[3][i2][o0 + oo];
        op[oo] = v * invL;
    }
}

// ---------------------------------------------------------------------------
extern "C" void kernel_launch(void* const* d_in, const int* in_sizes, int n_in,
                              void* d_out, int out_size, void* d_ws, size_t ws_size,
                              hipStream_t stream)
{
    const float* h   = (const float*)d_in[0];   // (B, N, FIN)
    const float* adj = (const float*)d_in[1];   // (N, N)
    const float* W   = (const float*)d_in[2];   // (H, FIN, FOUT)
    const float* a   = (const float*)d_in[3];   // (H, 2*FOUT, 1)
    float* out = (float*)d_out;                 // (B, N, H*FOUT)

    // workspace: [0, 4MB) swizzled f16 ht ; then es, ed (128 KB each)
    _Float16* ht16 = (_Float16*)d_ws;
    float* es = (float*)((char*)d_ws + (size_t)GAT_B * GAT_H * GAT_N * GAT_FOUT * 2);
    float* ed = es + (size_t)GAT_B * GAT_H * GAT_N;

    gat_stage1<<<dim3(GAT_N, GAT_B), 256, 0, stream>>>(h, W, a, ht16, es, ed);
    gat_stage2<<<dim3(GAT_N / 16, GAT_H, GAT_B), 128, 0, stream>>>(adj, ht16, es, ed, out);
}
